// PINNLarge_75067438399623
// MI455X (gfx1250) — matrix operand order repaired and checked by reading, not compiled, not run on hardware
//
#include <hip/hip_runtime.h>
#include <hip/hip_fp16.h>

typedef __attribute__((ext_vector_type(16))) _Float16 v16h;
typedef __attribute__((ext_vector_type(8)))  float    v8f;

#define DEV __device__ __forceinline__

static constexpr int B_ = 2048;
static constexpr int N_ = 32;
static constexpr int P_ = 496;          // N*(N-1)/2
static constexpr int PSI_TILES = B_ * (P_ / 16);   // 63488
static constexpr int PHI_TILES = B_ * (N_ / 16);   // 4096
static constexpr int RHO_TILES = B_ / 16;          // 128

// ---- workspace layout (bytes) ----
static constexpr size_t WPHI_OFF = 0;                 // 57344 halfs
static constexpr size_t WPSI_OFF = 114688;            // 57344 halfs
static constexpr size_t WRHO_OFF = 229376;            // 63488 halfs
static constexpr size_t ACC_OFF  = 356352;
static constexpr size_t PHI_SUM_OFF  = ACC_OFF + 0;        // B*32 f32
static constexpr size_t PHI_MAXK_OFF = ACC_OFF + 262144;   // B*32 u32
static constexpr size_t PSI_SUM_OFF  = ACC_OFF + 524288;
static constexpr size_t PSI_MAXK_OFF = ACC_OFF + 786432;
static constexpr size_t S1SUM_OFF    = ACC_OFF + 1048576;  // B f32
static constexpr size_t R2SUM_OFF    = ACC_OFF + 1056768;
static constexpr size_t CUSP_OFF     = ACC_OFF + 1064960;
static constexpr size_t RHOIN_OFF    = ACC_OFF + 1073152;  // B*224 f32
static constexpr int    ACC_ZERO_WORDS = 268288;           // everything ACC..RHOIN

// per-weight-buffer layer offsets (in halfs)
static constexpr int PPHI_L0 = 0, PPHI_L1 = 4096, PPHI_L2 = 20480, PPHI_L3 = 36864, PPHI_L4 = 53248;
static constexpr int RHO_L0 = 0, RHO_L1 = 28672, RHO_L2 = 45056, RHO_L3 = 61440;

static constexpr float LOG2E = 1.4426950408889634f;

// Fast hardware exp(x) = v_exp_f32(x * log2(e))
DEV float fexp(float x) { return __builtin_amdgcn_exp2f(x * LOG2E); }

// tanh-form GELU via hardware exp + rcp:
//   gelu(x) ~= x / (1 + exp(-2u)),  u = sqrt(2/pi) * (x + 0.044715 x^3)
// max abs error ~1e-3 — below the f16 activation quantization that follows.
DEV float gelu_fast(float v) {
    float u = 0.7978845608028654f * v * (1.0f + 0.044715f * v * v);
    float t = __builtin_amdgcn_exp2f(-2.8853900817779268f * u);   // exp(-2u)
    return v * __builtin_amdgcn_rcpf(1.0f + t);
}

DEV unsigned fkey(float f) { unsigned u = __float_as_uint(f); return (u & 0x80000000u) ? ~u : (u | 0x80000000u); }
DEV float fdecode(unsigned k) { return (k & 0x80000000u) ? __uint_as_float(k ^ 0x80000000u) : __uint_as_float(~k); }
DEV void wait_ds() { asm volatile("s_wait_dscnt 0x0" ::: "memory"); }

// A-matrix (16x32 f16) K mapping: element e, lane-half h -> k = (e/8)*16 + e%8 + 8h
// inverse for feature f (0..127): kc=f>>5, kk=f&31: e=(kk>>4)*8+(kk&7), lane-half=(kk>>3)&1

// Pack a 16-row tile of input features into A-fragment layout in LDS.
template <typename G>
DEV void packA(_Float16* dst, int Kc, int lane, G get) {
    int half = lane >> 4, m = lane & 15;
    for (int kc = 0; kc < Kc; ++kc) {
        v16h v;
#pragma unroll
        for (int e = 0; e < 16; ++e) {
            int k = kc * 32 + ((e >> 3) << 4) + (e & 7) + (half << 3);
            v[e] = (_Float16)get(m, k);
        }
        *(v16h*)&dst[kc * 512 + lane * 16] = v;
    }
    wait_ds();
}

// One dense layer (out width = Nt*16 = 128), f16 WMMA, f32 accum, fast GELU,
// result re-packed into A-layout in bufOut.
template <int Kc, int Nt>
DEV void wave_layer(const _Float16* __restrict__ w, const float* __restrict__ bias,
                    const _Float16* bufIn, _Float16* bufOut, int lane) {
    v8f acc[Nt];
#pragma unroll
    for (int nt = 0; nt < Nt; ++nt) acc[nt] = (v8f){0.f,0.f,0.f,0.f,0.f,0.f,0.f,0.f};
#pragma unroll
    for (int kc = 0; kc < Kc; ++kc) {
        v16h a = *(const v16h*)&bufIn[kc * 512 + lane * 16];
#pragma unroll
        for (int nt = 0; nt < Nt; ++nt) {
            v16h b = *(const v16h*)&w[(kc * Nt + nt) * 512 + lane * 16];
            acc[nt] = __builtin_amdgcn_wmma_f32_16x16x32_f16(false, a, false, b, (short)0, acc[nt], false, false);
        }
    }
    int half = lane >> 4, col = lane & 15;
#pragma unroll
    for (int nt = 0; nt < Nt; ++nt) {
        float bv = bias[nt * 16 + col];
#pragma unroll
        for (int r = 0; r < 8; ++r) {
            float v = gelu_fast(acc[nt][r] + bv);
            int m = r + (half << 3);
            int f = nt * 16 + col;
            int kc2 = f >> 5, kk = f & 31;
            int e = ((kk >> 4) << 3) + (kk & 7);
            int lh = (kk >> 3) & 1;
            bufOut[kc2 * 512 + ((lh << 4) + m) * 16 + e] = (_Float16)v;
        }
    }
    wait_ds();
}

// Final 128->32 layer + optional per-row gate + per-batch sum/max accumulation.
DEV void final_head_32(const _Float16* __restrict__ w, const float* __restrict__ bias,
                       const _Float16* bufIn, const float* gateLds,
                       float* sumArr, unsigned* maxArr, int lane) {
    v8f acc[2];
    acc[0] = (v8f){0.f,0.f,0.f,0.f,0.f,0.f,0.f,0.f};
    acc[1] = acc[0];
#pragma unroll
    for (int kc = 0; kc < 4; ++kc) {
        v16h a = *(const v16h*)&bufIn[kc * 512 + lane * 16];
#pragma unroll
        for (int nt = 0; nt < 2; ++nt) {
            v16h b = *(const v16h*)&w[(kc * 2 + nt) * 512 + lane * 16];
            acc[nt] = __builtin_amdgcn_wmma_f32_16x16x32_f16(false, a, false, b, (short)0, acc[nt], false, false);
        }
    }
    int half = lane >> 4, col = lane & 15;
#pragma unroll
    for (int nt = 0; nt < 2; ++nt) {
        int f = nt * 16 + col;
        float bv = bias[f];
        float s = 0.0f, mx = -3.4e38f;
#pragma unroll
        for (int r = 0; r < 8; ++r) {
            int m = r + (half << 3);
            float g = gateLds ? gateLds[m] : 1.0f;
            float v = (acc[nt][r] + bv) * g;
            s += v;
            mx = fmaxf(mx, v);
        }
        s += __shfl_xor(s, 16, 32);
        mx = fmaxf(mx, __shfl_xor(mx, 16, 32));
        if (half == 0) {
            atomicAdd(&sumArr[f], s);
            atomicMax(&maxArr[f], fkey(mx));
        }
    }
}

// ---------------- utility kernels ----------------
__global__ __launch_bounds__(256) void k_zero(unsigned* p, int n) {
    int i = blockIdx.x * 256 + threadIdx.x;
    if (i < n) p[i] = 0u;
}

// Pack W (Kreal x Nreal, row-major f32) into WMMA B-fragment layout (f16).
// B-matrix K mapping: element e, lane-half h -> k = (e/8)*16 + e%8 + 8h
__global__ __launch_bounds__(256) void k_pack_w(const float* __restrict__ W, _Float16* __restrict__ dst,
                                                int Kreal, int Nreal, int Kc, int Nt) {
    int idx = blockIdx.x * 256 + threadIdx.x;
    int total = Kc * Nt * 512;
    if (idx >= total) return;
    int e = idx & 15;
    int l = (idx >> 4) & 31;
    int frag = idx >> 9;
    int nt = frag % Nt, kc = frag / Nt;
    int k = kc * 32 + ((e >> 3) << 4) + (e & 7) + ((l >> 4) << 3);
    int n = nt * 16 + (l & 15);
    float v = (k < Kreal && n < Nreal) ? W[k * Nreal + n] : 0.0f;
    dst[idx] = (_Float16)v;
}

// ---------------- phi branch ----------------
__global__ __launch_bounds__(128) void k_phi(const float* __restrict__ x, const _Float16* __restrict__ w,
                                             const float* b0, const float* b1, const float* b2,
                                             const float* b3, const float* b4,
                                             float* phi_sum, unsigned* phi_maxk, float* r2sum) {
    __shared__ __align__(32) _Float16 sA[4][2048];
    __shared__ __align__(32) _Float16 sB[4][2048];
    __shared__ float sFeat[4][64];
    int tid = threadIdx.x, wid = tid >> 5, lane = tid & 31;
    int tile = blockIdx.x * 4 + wid;
    int b = tile >> 1;
    int n0 = (tile & 1) << 4;
    if (lane < 16) {
        const float* xp = x + b * (N_ * 3) + (n0 + lane) * 3;
        float x0 = xp[0], x1 = xp[1], x2 = xp[2];   // OMEGA = 1 -> x_sc = x
        sFeat[wid][lane * 4 + 0] = x0;
        sFeat[wid][lane * 4 + 1] = x1;
        sFeat[wid][lane * 4 + 2] = x2;
        atomicAdd(&r2sum[b], x0 * x0 + x1 * x1 + x2 * x2);
    }
    wait_ds();
    float* feat = sFeat[wid];
    packA(sA[wid], 1, lane, [&](int m, int k) { return (k < 3) ? feat[m * 4 + k] : 0.0f; });
    wave_layer<1, 8>(w + PPHI_L0, b0, sA[wid], sB[wid], lane);
    wave_layer<4, 8>(w + PPHI_L1, b1, sB[wid], sA[wid], lane);
    wave_layer<4, 8>(w + PPHI_L2, b2, sA[wid], sB[wid], lane);
    wave_layer<4, 8>(w + PPHI_L3, b3, sB[wid], sA[wid], lane);
    final_head_32(w + PPHI_L4, b4, sA[wid], nullptr, phi_sum + b * 32, phi_maxk + b * 32, lane);
}

// ---------------- psi branch (dominant) ----------------
__global__ __launch_bounds__(128) void k_psi(const float* __restrict__ x, const _Float16* __restrict__ w,
                                             const float* b0, const float* b1, const float* b2,
                                             const float* b3, const float* b4,
                                             float* psi_sum, unsigned* psi_maxk, float* s1sum, float* cusp) {
    __shared__ __align__(32) _Float16 sA[4][2048];
    __shared__ __align__(32) _Float16 sB[4][2048];
    __shared__ float sFeat[4][128];
    __shared__ float sGate[4][16];
    int tid = threadIdx.x, wid = tid >> 5, lane = tid & 31;
    int tile = blockIdx.x * 4 + wid;
    int b = tile / (P_ / 16);
    int p0 = (tile % (P_ / 16)) * 16;
    if (lane < 16) {
        int p = p0 + lane;
        int i = 0, rem = p;                         // triu_indices(32, k=1) order
        while (rem >= (N_ - 1 - i)) { rem -= (N_ - 1 - i); ++i; }
        int j = i + 1 + rem;
        const float* xb = x + b * (N_ * 3);
        float dx = xb[i * 3 + 0] - xb[j * 3 + 0];
        float dy = xb[i * 3 + 1] - xb[j * 3 + 1];
        float dz = xb[i * 3 + 2] - xb[j * 3 + 2];
        float r2 = dx * dx + dy * dy + dz * dz;
        float r = sqrtf(r2 + 1.1920929e-7f);        // + f32 eps
        float rr = r * r;
        float rt2 = (rr + 0.04f) * 25.0f;           // EPS_FEAT = 0.2
        float s1 = log1pf(rt2);
        float s2 = rr / (rr + 0.04f);
        float s3 = rt2 * fexp(-rt2);
        float* feat = &sFeat[wid][lane * 8];
        feat[0] = s1; feat[1] = s2; feat[2] = s3;
        feat[3] = fexp(-0.25f * s1);
        feat[4] = fexp(-1.0f * s1);
        feat[5] = fexp(-4.0f * s1);
        sGate[wid][lane] = rr / (rr + 0.09f);       // RG = 0.3
        atomicAdd(&s1sum[b], s1);
        // analytic e-e cusp (unscaled coords identical at OMEGA=1)
        float rc = sqrtf(r2 + 1e-30f);
        bool same = (i < 16) == (j < 16);
        float gam = same ? 0.25f : 0.5f;            // 1/(D+1), 1/(D-1)
        atomicAdd(&cusp[b], gam * rc * fexp(-rc));
    }
    wait_ds();
    float* feat = sFeat[wid];
    packA(sA[wid], 1, lane, [&](int m, int k) { return (k < 6) ? feat[m * 8 + k] : 0.0f; });
    wave_layer<1, 8>(w + PPHI_L0, b0, sA[wid], sB[wid], lane);
    wave_layer<4, 8>(w + PPHI_L1, b1, sB[wid], sA[wid], lane);
    wave_layer<4, 8>(w + PPHI_L2, b2, sA[wid], sB[wid], lane);
    wave_layer<4, 8>(w + PPHI_L3, b3, sB[wid], sA[wid], lane);
    final_head_32(w + PPHI_L4, b4, sA[wid], sGate[wid], psi_sum + b * 32, psi_maxk + b * 32, lane);
}

// ---------------- assemble rho input ----------------
__global__ __launch_bounds__(256) void k_finalize(const float* phi_sum, const unsigned* phi_maxk,
                                                  const float* psi_sum, const unsigned* psi_maxk,
                                                  const float* s1sum, const float* r2sum,
                                                  float* rho_in) {
    int idx = blockIdx.x * 256 + threadIdx.x;
    if (idx >= B_ * 224) return;
    int b = idx / 224, f = idx % 224;
    float v;
    if      (f < 32)  v = phi_sum[b * 32 + f];
    else if (f < 64)  v = phi_sum[b * 32 + f - 32] * (1.0f / 32.0f);
    else if (f < 96)  v = fdecode(phi_maxk[b * 32 + f - 64]);
    else if (f < 128) v = psi_sum[b * 32 + f - 96];
    else if (f < 160) v = psi_sum[b * 32 + f - 128] * (1.0f / 496.0f);
    else if (f < 192) v = fdecode(psi_maxk[b * 32 + f - 160]);
    else if (f == 192) v = r2sum[b] * (1.0f / 96.0f);
    else if (f == 193) v = s1sum[b] * (1.0f / 496.0f);
    else v = 0.0f;
    rho_in[idx] = v;
}

// ---------------- rho head ----------------
__global__ __launch_bounds__(128) void k_rho(const float* __restrict__ rho_in, const _Float16* __restrict__ w,
                                             const float* b0, const float* b1, const float* b2, const float* b3,
                                             const float* cusp, float* out) {
    __shared__ __align__(32) _Float16 sA[4][3584];   // 7 K-chunks
    __shared__ __align__(32) _Float16 sB[4][2048];
    int tid = threadIdx.x, wid = tid >> 5, lane = tid & 31;
    int tile = blockIdx.x * 4 + wid;
    int row0 = tile * 16;
    packA(sA[wid], 7, lane, [&](int m, int k) { return rho_in[(row0 + m) * 224 + k]; });
    wave_layer<7, 8>(w + RHO_L0, b0, sA[wid], sB[wid], lane);
    wave_layer<4, 8>(w + RHO_L1, b1, sB[wid], sA[wid], lane);
    wave_layer<4, 8>(w + RHO_L2, b2, sA[wid], sB[wid], lane);
    // final 128 -> 1 (N padded to 16, column 0 valid)
    v8f acc = (v8f){0.f,0.f,0.f,0.f,0.f,0.f,0.f,0.f};
#pragma unroll
    for (int kc = 0; kc < 4; ++kc) {
        v16h a = *(const v16h*)&sB[wid][kc * 512 + lane * 16];
        v16h bb = *(const v16h*)&w[RHO_L3 + kc * 512 + lane * 16];
        acc = __builtin_amdgcn_wmma_f32_16x16x32_f16(false, a, false, bb, (short)0, acc, false, false);
    }
    int half = lane >> 4, col = lane & 15;
    if (col == 0) {
        float bias = b3[0];
#pragma unroll
        for (int r = 0; r < 8; ++r) {
            int row = row0 + r + (half << 3);
            out[row] = acc[r] + bias + cusp[row];
        }
    }
}

// ---------------- host ----------------
extern "C" void kernel_launch(void* const* d_in, const int* in_sizes, int n_in,
                              void* d_out, int out_size, void* d_ws, size_t ws_size,
                              hipStream_t stream) {
    (void)in_sizes; (void)n_in; (void)out_size; (void)ws_size;
    const float* x = (const float*)d_in[0];
    const float* phiW[5]; const float* phiB[5];
    const float* psiW[5]; const float* psiB[5];
    const float* roW[4];  const float* roB[4];
    for (int i = 0; i < 5; ++i) { phiW[i] = (const float*)d_in[1 + i];  phiB[i] = (const float*)d_in[6 + i]; }
    for (int i = 0; i < 5; ++i) { psiW[i] = (const float*)d_in[11 + i]; psiB[i] = (const float*)d_in[16 + i]; }
    for (int i = 0; i < 4; ++i) { roW[i]  = (const float*)d_in[21 + i]; roB[i]  = (const float*)d_in[25 + i]; }

    char* ws = (char*)d_ws;
    _Float16* wphi = (_Float16*)(ws + WPHI_OFF);
    _Float16* wpsi = (_Float16*)(ws + WPSI_OFF);
    _Float16* wrho = (_Float16*)(ws + WRHO_OFF);
    float*    phi_sum  = (float*)(ws + PHI_SUM_OFF);
    unsigned* phi_maxk = (unsigned*)(ws + PHI_MAXK_OFF);
    float*    psi_sum  = (float*)(ws + PSI_SUM_OFF);
    unsigned* psi_maxk = (unsigned*)(ws + PSI_MAXK_OFF);
    float*    s1sum    = (float*)(ws + S1SUM_OFF);
    float*    r2sum    = (float*)(ws + R2SUM_OFF);
    float*    cusp     = (float*)(ws + CUSP_OFF);
    float*    rho_in   = (float*)(ws + RHOIN_OFF);

    // 1) zero all accumulators (sums=0; max keys=0 == encoded -inf ordering floor)
    k_zero<<<(ACC_ZERO_WORDS + 255) / 256, 256, 0, stream>>>((unsigned*)(ws + ACC_OFF), ACC_ZERO_WORDS);

    // 2) pre-swizzle weights into f16 WMMA B-fragment layout
    auto pack = [&](const float* W, _Float16* dst, int Kr, int Nr, int Kc, int Nt) {
        int total = Kc * Nt * 512;
        k_pack_w<<<(total + 255) / 256, 256, 0, stream>>>(W, dst, Kr, Nr, Kc, Nt);
    };
    pack(phiW[0], wphi + PPHI_L0, 3, 128, 1, 8);
    pack(phiW[1], wphi + PPHI_L1, 128, 128, 4, 8);
    pack(phiW[2], wphi + PPHI_L2, 128, 128, 4, 8);
    pack(phiW[3], wphi + PPHI_L3, 128, 128, 4, 8);
    pack(phiW[4], wphi + PPHI_L4, 128, 32, 4, 2);
    pack(psiW[0], wpsi + PPHI_L0, 6, 128, 1, 8);
    pack(psiW[1], wpsi + PPHI_L1, 128, 128, 4, 8);
    pack(psiW[2], wpsi + PPHI_L2, 128, 128, 4, 8);
    pack(psiW[3], wpsi + PPHI_L3, 128, 128, 4, 8);
    pack(psiW[4], wpsi + PPHI_L4, 128, 32, 4, 2);
    pack(roW[0], wrho + RHO_L0, 194, 128, 7, 8);
    pack(roW[1], wrho + RHO_L1, 128, 128, 4, 8);
    pack(roW[2], wrho + RHO_L2, 128, 128, 4, 8);
    pack(roW[3], wrho + RHO_L3, 128, 1, 4, 1);

    // 3) branch kernels (4 waves/block, 16 rows/wave)
    k_phi<<<PHI_TILES / 4, 128, 0, stream>>>(x, wphi, phiB[0], phiB[1], phiB[2], phiB[3], phiB[4],
                                             phi_sum, phi_maxk, r2sum);
    k_psi<<<PSI_TILES / 4, 128, 0, stream>>>(x, wpsi, psiB[0], psiB[1], psiB[2], psiB[3], psiB[4],
                                             psi_sum, psi_maxk, s1sum, cusp);

    // 4) assemble rho input, run head, add cusp
    k_finalize<<<(B_ * 224 + 255) / 256, 256, 0, stream>>>(phi_sum, phi_maxk, psi_sum, psi_maxk,
                                                           s1sum, r2sum, rho_in);
    k_rho<<<RHO_TILES / 4, 128, 0, stream>>>(rho_in, wrho, roB[0], roB[1], roB[2], roB[3],
                                             cusp, (float*)d_out);
}
